// GraphAttentionNetwork_30915174597232
// MI455X (gfx1250) — matrix operand order repaired and checked by reading, not compiled, not run on hardware
//
#include <hip/hip_runtime.h>
#include <hip/hip_bf16.h>

// ---------------------------------------------------------------------------
// GAT on gfx1250: bitmask-compressed adjacency + tile-skipping flash attention
// with v_wmma_f32_16x16x32_bf16. WMMA operands staged in LDS in register order
// (ds_load_b128 pairs); register-resident online softmax with wave shuffles;
// double-buffered GEMM.
// ---------------------------------------------------------------------------

#define GAT_N       4096
#define GAT_INDIM   256
#define GAT_HID     64
#define GAT_HEADS   8
#define GAT_OUTDIM  64
#define GAT_ALPHA   0.2f
#define GAT_NEGBIG  (-1.0e30f)

typedef __attribute__((ext_vector_type(16))) __bf16 v16bf;
typedef __attribute__((ext_vector_type(8)))  float  v8f;

static __device__ __forceinline__ unsigned short f2bfu(float f) {
    // round-to-nearest-even f32 -> bf16
    unsigned u = __builtin_bit_cast(unsigned, f);
    unsigned r = u + 0x7FFFu + ((u >> 16) & 1u);
    return (unsigned short)(r >> 16);
}

// WMMA register-order index helpers (ISA 7.12.2, wave32):
//   A (16x32 MxK):  lane = M + 16*((K>>3)&1),  elem = (K&7) + 8*(K>>4)
//   B (32x16 KxN):  lane = N + 16*((K>>3)&1),  elem = (K&7) + 8*(K>>4)
static __device__ __forceinline__ int aLane(int m, int k) { return m + 16 * ((k >> 3) & 1); }
static __device__ __forceinline__ int aElem(int k)        { return (k & 7) + 8 * (k >> 4); }
static __device__ __forceinline__ int bLane(int n, int k) { return (n & 15) + 16 * ((k >> 3) & 1); }

// ---------------------------------------------------------------------------
// Kernel 1: adjacency -> bitmask via wave32 ballot (64MB read once).
// ---------------------------------------------------------------------------
__global__ __launch_bounds__(256) void gat_build_mask(
    const float* __restrict__ adj, unsigned int* __restrict__ maskW, int n)
{
    const int wavesPerBlock = blockDim.x >> 5;
    const int gw   = blockIdx.x * wavesPerBlock + (threadIdx.x >> 5);
    const int lane = threadIdx.x & 31;
    const int wordsPerRow = n >> 5;
    const int row = gw / wordsPerRow;
    const int wi  = gw - row * wordsPerRow;
    if (row >= n) return;
    float v = adj[(long)row * n + (wi << 5) + lane];
    unsigned int m = __builtin_amdgcn_ballot_w32(v > 0.0f);
    if (lane == 0) maskW[gw] = m;
}

// ---------------------------------------------------------------------------
// Kernel 1b: tile occupancy = OR of 16 row mask-words per 16-row block.
// ---------------------------------------------------------------------------
__global__ __launch_bounds__(128) void gat_build_occ(
    const unsigned int* __restrict__ maskW, unsigned int* __restrict__ occ, int n)
{
    const int wordsPerRow = n >> 5;
    const int rb = blockIdx.x;                 // row block (16 rows)
    for (int j = threadIdx.x; j < wordsPerRow; j += blockDim.x) {
        unsigned int a = 0;
        for (int i = 0; i < 16; ++i)
            a |= maskW[(rb * 16 + i) * wordsPerRow + j];
        occ[rb * wordsPerRow + j] = a;
    }
}

// ---------------------------------------------------------------------------
// Kernel 2: per-head feature GEMM + attention scores (double-buffered LDS).
//   Wh[N,64] = X[N,F] @ W[F,64]   (bf16 WMMA, f32 accumulate)
//   f1[N] = Wh . a1 ; f2[N] = Wh . a2
// Block = 128 threads (4 waves); wave w owns output columns [16w,16w+16).
// ---------------------------------------------------------------------------
__global__ __launch_bounds__(128) void gat_gemm_score(
    const float* __restrict__ X, int ldX,
    const float* __restrict__ W,  long wHeadStride,
    const float* __restrict__ A1, const float* __restrict__ A2, long aHeadStride,
    unsigned short* __restrict__ WhBf, long whHeadStride,
    float* __restrict__ F1, float* __restrict__ F2, long fHeadStride,
    int F)
{
    const int tid  = threadIdx.x;
    const int head = blockIdx.y;
    const int rowBase = blockIdx.x * 16;

    const float* Wp = W  + (long)head * wHeadStride;
    const float* a1 = A1 + (long)head * aHeadStride;
    const float* a2 = A2 + (long)head * aHeadStride;
    unsigned short* wh = WhBf + (long)head * whHeadStride;
    float* f1 = F1 + (long)head * fHeadStride;
    float* f2 = F2 + (long)head * fHeadStride;

    // Ping-pong operand tiles in WMMA register order:
    __shared__ v16bf sAw[2][32];
    __shared__ v16bf sBw[2][128];
    __shared__ float sC[16 * 64];

    unsigned short* sAu = (unsigned short*)sAw;   // buffer b at offset b*512
    unsigned short* sBu = (unsigned short*)sBw;   // buffer b at offset b*2048

    const int lane = tid & 31;
    const int wv   = tid >> 5;
    const int hl   = lane >> 4;
    const int lm   = lane & 15;

    // stage one 32-wide K tile into buffer b (int indexing: all offsets < 2^31)
    auto stage = [&](int k0, int b) {
#pragma unroll
        for (int ii = 0; ii < 4; ++ii) {
            int i = tid + ii * 128;
            int m = i >> 5, k = i & 31;
            sAu[b * 512 + aLane(m, k) * 16 + aElem(k)] =
                f2bfu(X[(rowBase + m) * ldX + k0 + k]);
        }
#pragma unroll
        for (int ii = 0; ii < 16; ++ii) {
            int i = tid + ii * 128;
            int k = i >> 6, n = i & 63;
            sBu[b * 2048 + ((n >> 4) * 32 + bLane(n, k)) * 16 + aElem(k)] =
                f2bfu(Wp[(k0 + k) * 64 + n]);
        }
    };

    v8f c = {};

    stage(0, 0);
    __syncthreads();
    int buf = 0;
    for (int k0 = 0; k0 < F; k0 += 32) {
        if (k0 + 32 < F) stage(k0 + 32, buf ^ 1);   // overlap with WMMA below
        v16bf a = sAw[buf][lane];                   // 2x ds_load_b128
        v16bf b = sBw[buf][wv * 32 + lane];         // 2x ds_load_b128
        c = __builtin_amdgcn_wmma_f32_16x16x32_bf16(
                false, a, false, b, (short)0, c, false, false);
        __syncthreads();
        buf ^= 1;
    }

    // D layout: vgpr r holds row (r + 8*half), col = lane&15 within wave tile
#pragma unroll
    for (int r = 0; r < 8; ++r) {
        int m   = r + 8 * hl;
        int col = wv * 16 + lm;
        float v = c[r];
        sC[m * 64 + col] = v;
        wh[(rowBase + m) * 64 + col] = f2bfu(v);
    }
    __syncthreads();

    if (tid < 16) {
        float s1 = 0.f, s2 = 0.f;
        for (int j = 0; j < 64; ++j) {
            float v = sC[tid * 64 + j];
            s1 += v * a1[j];
            s2 += v * a2[j];
        }
        f1[rowBase + tid] = s1;
        f2[rowBase + tid] = s2;
    }
}

// ---------------------------------------------------------------------------
// Kernel 3: tile-skipping flash attention, register-resident online softmax.
//   e[n,m] = leakyrelu(f1[n]+f2[m]) where adj-bit set, else -inf
//   out[n,:] = softmax_m(e) @ Wh      (P@V via bf16 WMMA)
// Block = 128 threads (4 waves), 16 rows, K-step = 32 columns.
// Row r is scored by lanes r*8..r*8+7 of one wave -> shuffle reductions,
// m/l state replicated in registers. Only sScale crosses waves via LDS.
// 2 barriers per processed tile; skipped tiles cost one LDS probe.
// mode 0: write f32 concat h[row, head*64+col]; mode 1: ELU -> d_out
// ---------------------------------------------------------------------------
__global__ __launch_bounds__(128) void gat_attn(
    const float* __restrict__ F1, const float* __restrict__ F2, long fHeadStride,
    const unsigned short* __restrict__ WhBf, long whHeadStride,
    const unsigned int* __restrict__ maskW,
    const unsigned int* __restrict__ occ,
    float* __restrict__ outp, int outLd, int n, int mode)
{
    const int tid  = threadIdx.x;
    const int head = blockIdx.y;
    const int rowBase = blockIdx.x * 16;
    const int wordsPerRow = n >> 5;

    const float* f1p = F1 + (long)head * fHeadStride;
    const float* f2p = F2 + (long)head * fHeadStride;
    const unsigned short* whp = WhBf + (long)head * whHeadStride;
    const int colOff = (mode == 0) ? head * 64 : 0;

    __shared__ v16bf sVw[4 * 32];    // V tile in B register order, [wave][lane]
    __shared__ v16bf sPw[32];        // P tile in A register order, [lane]
    __shared__ float sO[16 * 64];
    __shared__ float sScale[16], sL[16];
    __shared__ unsigned int sOcc[GAT_N / 32];   // 128 words

    unsigned short* sVu = (unsigned short*)sVw;
    unsigned short* sPu = (unsigned short*)sPw;

    const int lane = tid & 31;
    const int wv   = tid >> 5;
    const int hl   = lane >> 4;
    const int lm   = lane & 15;

    const int r = tid >> 3;       // row 0..15 this thread scores
    const int t = tid & 7;        // 8 lanes per row (consecutive, same wave)
    // P-store target: this thread's 4 columns land in 4 consecutive elem slots
    const int pLane = r + 16 * ((t >> 1) & 1);
    const int pEl   = 4 * (t & 1) + 8 * (t >> 2);

    // preload per-row-block occupancy words into LDS (skip probe = 1 LDS read)
    for (int i = tid; i < wordsPerRow; i += 128)
        sOcc[i] = occ[blockIdx.x * wordsPerRow + i];

    const float f1r = f1p[rowBase + r];
    const unsigned int* mrow = maskW + (rowBase + r) * wordsPerRow;

    float mRow = GAT_NEGBIG;      // replicated online-softmax state (x8 lanes)
    float lRow = 0.f;
    v8f c = {};

    __syncthreads();              // sOcc visible

    for (int j0 = 0; j0 < n; j0 += 32) {
        if (sOcc[j0 >> 5] == 0) continue;      // uniform, barrier-free skip

        // ---- scores (pure registers: global mask word + float4 f2) ----
        const unsigned int mword = mrow[j0 >> 5];
        const float4 f2v = *(const float4*)(f2p + j0 + t * 4);
        const float f2a[4] = { f2v.x, f2v.y, f2v.z, f2v.w };
        float ev[4];
        float tmax = GAT_NEGBIG;
#pragma unroll
        for (int i = 0; i < 4; ++i) {
            int cc = t * 4 + i;
            float e = GAT_NEGBIG;
            if ((mword >> cc) & 1u) {
                e = f1r + f2a[i];
                e = (e > 0.f) ? e : GAT_ALPHA * e;
            }
            ev[i] = e;
            tmax = fmaxf(tmax, e);
        }
        // row-tile max across the 8 scoring lanes (intra-wave butterfly)
#pragma unroll
        for (int s = 1; s <= 4; s <<= 1)
            tmax = fmaxf(tmax, __shfl_xor(tmax, s, 32));

        const float mNew  = fmaxf(mRow, tmax);
        const float scale = __expf(mRow - mNew);   // (-big)-(-big)=0 -> 1
        mRow = mNew;

        float p[4];
        float ps = 0.f;
#pragma unroll
        for (int i = 0; i < 4; ++i) {
            int cc = t * 4 + i;
            p[i] = ((mword >> cc) & 1u) ? __expf(ev[i] - mNew) : 0.f;
            ps += p[i];
        }
#pragma unroll
        for (int s = 1; s <= 4; s <<= 1)
            ps += __shfl_xor(ps, s, 32);
        lRow = lRow * scale + ps;

        __syncthreads();   // previous tile's WMMA is done with sPw/sVw

        // ---- store P (packed b64, A register order) + cross-wave scale ----
        unsigned int u0 = (unsigned)f2bfu(p[0]) | ((unsigned)f2bfu(p[1]) << 16);
        unsigned int u1 = (unsigned)f2bfu(p[2]) | ((unsigned)f2bfu(p[3]) << 16);
        *(unsigned long long*)(sPu + pLane * 16 + pEl) =
            (unsigned long long)u0 | ((unsigned long long)u1 << 32);
        if (t == 0) sScale[r] = scale;

        // ---- stage V tile (B register order) ----
        {
            const unsigned short* vsrc = whp + j0 * 64;
#pragma unroll
            for (int ii = 0; ii < 16; ++ii) {
                int i = tid + ii * 128;
                int k = i >> 6, nn = i & 63;
                sVu[((nn >> 4) * 32 + bLane(nn, k)) * 16 + aElem(k)] = vsrc[i];
            }
        }
        __syncthreads();

        // ---- rescale accumulator and add P@V (convergent) ----
        v16bf a = sPw[lane];                  // 2x ds_load_b128
        v16bf b = sVw[wv * 32 + lane];        // 2x ds_load_b128
#pragma unroll
        for (int q = 0; q < 8; ++q) c[q] *= sScale[q + 8 * hl];
        c = __builtin_amdgcn_wmma_f32_16x16x32_bf16(
                false, a, false, b, (short)0, c, false, false);
    }

    // ---- epilogue: normalize and write ----
#pragma unroll
    for (int q = 0; q < 8; ++q)
        sO[(q + 8 * hl) * 64 + wv * 16 + lm] = c[q];
    if (t == 0) sL[r] = lRow;
    __syncthreads();

    for (int i = tid; i < 16 * 64; i += 128) {
        int m = i >> 6, col = i & 63;
        float l = sL[m];
        float v = (l > 0.f) ? sO[i] / l : 0.f;
        int row = rowBase + m;
        if (mode == 0) {
            outp[row * outLd + colOff + col] = v;
        } else {
            float o = (v > 0.f) ? v : (__expf(v) - 1.f);   // ELU(alpha=1)
            outp[row * outLd + col] = o;
        }
    }
}

// ---------------------------------------------------------------------------
// Host-side launch
// ---------------------------------------------------------------------------
extern "C" void kernel_launch(void* const* d_in, const int* in_sizes, int n_in,
                              void* d_out, int out_size, void* d_ws, size_t ws_size,
                              hipStream_t stream) {
    (void)in_sizes; (void)n_in; (void)out_size; (void)ws_size;

    const float* x    = (const float*)d_in[0];   // [4096,256]
    const float* adj  = (const float*)d_in[1];   // [4096,4096]
    const float* W_h  = (const float*)d_in[2];   // [8,256,64]
    const float* a1_h = (const float*)d_in[3];   // [8,64]
    const float* a2_h = (const float*)d_in[4];   // [8,64]
    const float* W_o  = (const float*)d_in[5];   // [512,64]
    const float* a1_o = (const float*)d_in[6];   // [64]
    const float* a2_o = (const float*)d_in[7];   // [64]
    float* out = (float*)d_out;                  // [4096,64]

    const int N = GAT_N;
    const int wordsPerRow = N / 32;

    // ---- workspace carve-up (aligned to 256B) ----
    char* base = (char*)d_ws;
    size_t off = 0;
    auto carve = [&](size_t bytes) {
        void* p = base + off;
        off += (bytes + 255) & ~(size_t)255;
        return p;
    };
    unsigned int*   maskW = (unsigned int*)  carve((size_t)N * wordsPerRow * 4);        // 2 MB
    unsigned int*   occW  = (unsigned int*)  carve((size_t)(N / 16) * wordsPerRow * 4); // 128 KB
    unsigned short* WhBf1 = (unsigned short*)carve((size_t)GAT_HEADS * N * 64 * 2);     // 4 MB
    float*          f1_1  = (float*)         carve((size_t)GAT_HEADS * N * 4);
    float*          f2_1  = (float*)         carve((size_t)GAT_HEADS * N * 4);
    float*          hcat  = (float*)         carve((size_t)N * GAT_HEADS * 64 * 4);     // 8 MB
    unsigned short* WhBf2 = (unsigned short*)carve((size_t)N * 64 * 2);
    float*          f1_2  = (float*)         carve((size_t)N * 4);
    float*          f2_2  = (float*)         carve((size_t)N * 4);

    // 1) adjacency -> bitmask (one wave per 32 columns)
    {
        int words = N * wordsPerRow;
        int wavesPerBlock = 256 / 32;
        int blocks = words / wavesPerBlock;
        gat_build_mask<<<blocks, 256, 0, stream>>>(adj, maskW, N);
    }
    // 1b) 16x32-tile occupancy for the skip path
    gat_build_occ<<<N / 16, 128, 0, stream>>>(maskW, occW, N);

    // 2) layer-1 feature GEMM + scores (8 heads)
    gat_gemm_score<<<dim3(N / 16, GAT_HEADS), 128, 0, stream>>>(
        x, GAT_INDIM,
        W_h, (long)GAT_INDIM * 64,
        a1_h, a2_h, 64,
        WhBf1, (long)N * 64,
        f1_1, f2_1, N,
        GAT_INDIM);

    // 3) layer-1 attention -> concat h [4096, 512]
    gat_attn<<<dim3(N / 16, GAT_HEADS), 128, 0, stream>>>(
        f1_1, f2_1, N,
        WhBf1, (long)N * 64,
        maskW, occW,
        hcat, GAT_HEADS * 64, N, /*mode=*/0);

    // 4) layer-2 feature GEMM + scores (1 head, F=512)
    gat_gemm_score<<<dim3(N / 16, 1), 128, 0, stream>>>(
        hcat, GAT_HEADS * 64,
        W_o, 0,
        a1_o, a2_o, 0,
        WhBf2, 0,
        f1_2, f2_2, 0,
        GAT_HEADS * 64);

    // 5) layer-2 attention + ELU -> d_out
    gat_attn<<<dim3(N / 16, 1), 128, 0, stream>>>(
        f1_2, f2_2, 0,
        WhBf2, 0,
        maskW, occW,
        out, GAT_OUTDIM, N, /*mode=*/1);
}
